// MultiHeadAttentionUnscaled_22479858827707
// MI455X (gfx1250) — compile-verified, matrix operand order checked by reading, not run on hardware
//
#include <hip/hip_runtime.h>
#include <hip/hip_bf16.h>

// ---------------------------------------------------------------------------
// MHA (unscaled) for MI455X / gfx1250, wave32, WMMA f16->f32 + TDM staging.
// B=2, S=2048, D=1024, H=16, DK=64.
// ---------------------------------------------------------------------------

#define B_  2
#define S_  2048
#define D_  1024
#define H_  16
#define DK_ 64
#define NEG_ (-1000000000.0f)

typedef __attribute__((ext_vector_type(16))) _Float16 v16h;
typedef __attribute__((ext_vector_type(8)))  _Float16 v8h;
typedef __attribute__((ext_vector_type(8)))  float    v8f;
typedef __attribute__((ext_vector_type(4)))  float    v4f;
typedef unsigned int u32x4 __attribute__((ext_vector_type(4)));
typedef int          i32x4 __attribute__((ext_vector_type(4)));
typedef int          i32x8 __attribute__((ext_vector_type(8)));

__device__ __forceinline__ v16h cat8(v8h lo, v8h hi) {
    return __builtin_shufflevector(lo, hi, 0,1,2,3,4,5,6,7,8,9,10,11,12,13,14,15);
}

__device__ __forceinline__ v8f wmma_f16(v16h a, v16h b, v8f c) {
    // D = A(16x32 f16) * B(32x16 f16) + C(16x16 f32)
    return __builtin_amdgcn_wmma_f32_16x16x32_f16(
        false, a, false, b, (short)0, c, false, false);
}

// ---------------------------------------------------------------------------
// Tensor Data Mover: 2D tile (f16 elements) global -> LDS, with LDS row
// padding of 8 halves per 64-half row (pad_interval=32 dwords -> code 4,
// pad_amount=4 dwords -> code 3), giving a 72-half row stride in LDS.
// Descriptor per CDNA5 ISA section 8.3/8.4 (count=1, type=2, data_size=2B).
// ---------------------------------------------------------------------------
__device__ __forceinline__ void tdm_load_tile_f16(
        unsigned ldsByteOff, const _Float16* gaddr,
        unsigned tensorDim0, unsigned tensorDim1,
        unsigned tileDim0, unsigned tileDim1,
        unsigned long long stride0) {
    const unsigned long long ga = (unsigned long long)(uintptr_t)gaddr;
    u32x4 g0;
    g0[0] = 1u;                                              // count=1 (user D#)
    g0[1] = ldsByteOff;                                      // lds_addr
    g0[2] = (unsigned)(ga & 0xffffffffull);                  // global_addr[31:0]
    g0[3] = (unsigned)((ga >> 32) & 0x1ffffffull)            // global_addr[56:32]
          | (2u << 30);                                      // type=2 ("image")
    i32x8 g1;
    g1[0] = (int)((1u << 16)      // data_size = 2 bytes
                | (1u << 20)      // pad_enable
                | (4u << 22)      // pad_interval: 32 dwords per row
                | (3u << 25));    // pad_amount:   4 dwords (8 halves)
    g1[1] = (int)((tensorDim0 & 0xffffu) << 16);             // tensor_dim0[15:0]
    g1[2] = (int)(((tensorDim0 >> 16) & 0xffffu)
                | ((tensorDim1 & 0xffffu) << 16));           // dim0 hi | dim1 lo
    g1[3] = (int)(((tensorDim1 >> 16) & 0xffffu)
                | ((tileDim0 & 0xffffu) << 16));             // dim1 hi | tile0
    g1[4] = (int)(tileDim1 & 0xffffu);                       // tile1, tile2=0
    g1[5] = (int)(stride0 & 0xffffffffull);                  // dim0_stride[31:0]
    g1[6] = (int)((stride0 >> 32) & 0xffffull);              // dim0_stride[47:32]
    g1[7] = 0;
    const i32x4 z4 = {0, 0, 0, 0};
#if defined(__clang_major__) && (__clang_major__ >= 23)
    const i32x8 z8 = {0, 0, 0, 0, 0, 0, 0, 0};
    __builtin_amdgcn_tensor_load_to_lds(g0, g1, z4, z4, z8, 0);
#else
    __builtin_amdgcn_tensor_load_to_lds(g0, g1, z4, z4, 0);
#endif
}

// ---------------------------------------------------------------------------
// Projection GEMM: Y = X @ W^T + bias (templated -> branch-free inner loop).
//   X: (4096, 1024), f32 (XHALF=0) or f16 (XHALF=1), row-major
//   W: (1024, 1024)  f32 row-major (out_feature, in_feature)
//   MODE 0: Y f16 [B][H][S][DK]; MODE 1: Y f16 [B][H][DK][S]; MODE 2: Y f32.
// Block: 256 threads = 8 waves; tile 64(M) x 128(N); K step 32.
// Wave w: M tile (w&3)*16; N tiles (w>>2)*64 + {0,16,32,48} -> 4 WMMA/K-step.
// ---------------------------------------------------------------------------
template <int XHALF, int MODE>
__global__ __launch_bounds__(256)
void mha_proj_wmma_kernel(const void* __restrict__ Xv,
                          const float* __restrict__ W,
                          const float* __restrict__ bias,
                          void* __restrict__ Y) {
    __shared__ _Float16 Xs[64 * 40];    // 64 rows x 32 k (stride 40: 16B rows)
    __shared__ _Float16 Ws[128 * 40];   // 128 rows x 32 k

    const int tid  = threadIdx.x;
    const int lane = tid & 31;
    const int w    = tid >> 5;
    const int mi   = w & 3;
    const int nj   = w >> 2;
    const int l15  = lane & 15;
    const int kc0  = (lane < 16) ? 0 : 8;
    const int hl   = (lane >= 16) ? 8 : 0;

    const int row0 = blockIdx.x * 64;
    const int col0 = blockIdx.y * 128;

    const int r  = tid >> 2;        // staging row 0..63
    const int c8 = (tid & 3) * 8;   // staging col group

    const float*    Xf = (const float*)Xv;
    const _Float16* Xh = (const _Float16*)Xv;

    v8f acc[4];
    #pragma unroll
    for (int s = 0; s < 4; ++s) acc[s] = (v8f){};

    for (int k0 = 0; k0 < D_; k0 += 32) {
        __syncthreads();
        // ---- stage X tile ----
        {
            v8h hv;
            if (XHALF) {
                hv = *(const v8h*)(Xh + (size_t)(row0 + r) * D_ + k0 + c8);
            } else {
                const float* p = Xf + (size_t)(row0 + r) * D_ + k0 + c8;
                v4f a0 = *(const v4f*)p;
                v4f a1 = *(const v4f*)(p + 4);
                hv[0] = (_Float16)a0[0]; hv[1] = (_Float16)a0[1];
                hv[2] = (_Float16)a0[2]; hv[3] = (_Float16)a0[3];
                hv[4] = (_Float16)a1[0]; hv[5] = (_Float16)a1[1];
                hv[6] = (_Float16)a1[2]; hv[7] = (_Float16)a1[3];
            }
            *(v8h*)(Xs + r * 40 + c8) = hv;
        }
        // ---- stage W tile (two rows per thread: r and r+64) ----
        #pragma unroll
        for (int half = 0; half < 2; ++half) {
            const int rw = r + half * 64;
            const float* p = W + (size_t)(col0 + rw) * D_ + k0 + c8;
            v4f a0 = *(const v4f*)p;
            v4f a1 = *(const v4f*)(p + 4);
            v8h hv;
            hv[0] = (_Float16)a0[0]; hv[1] = (_Float16)a0[1];
            hv[2] = (_Float16)a0[2]; hv[3] = (_Float16)a0[3];
            hv[4] = (_Float16)a1[0]; hv[5] = (_Float16)a1[1];
            hv[6] = (_Float16)a1[2]; hv[7] = (_Float16)a1[3];
            *(v8h*)(Ws + rw * 40 + c8) = hv;
        }
        __syncthreads();

        // ---- fragments + 4 WMMAs ----
        const int ra = mi * 16 + l15;
        v16h af = cat8(*(const v8h*)(Xs + ra * 40 + kc0),
                       *(const v8h*)(Xs + ra * 40 + kc0 + 16));
        #pragma unroll
        for (int s = 0; s < 4; ++s) {
            const int rb = nj * 64 + s * 16 + l15;
            v16h bf = cat8(*(const v8h*)(Ws + rb * 40 + kc0),
                           *(const v8h*)(Ws + rb * 40 + kc0 + 16));
            acc[s] = wmma_f16(af, bf, acc[s]);
        }
    }

    // ---- epilogue: bias + scatter store ----
    #pragma unroll
    for (int s = 0; s < 4; ++s) {
        const int gn = col0 + nj * 64 + s * 16 + l15;
        const float bv = bias[gn];
        #pragma unroll
        for (int i = 0; i < 8; ++i) {
            const int gm = row0 + mi * 16 + i + hl;
            const float val = acc[s][i] + bv;
            if (MODE == 2) {
                ((float*)Y)[(size_t)gm * D_ + gn] = val;
            } else {
                const int b = gm >> 11;        // / S_
                const int ss = gm & (S_ - 1);
                const int h = gn >> 6;         // / DK_
                const int d = gn & (DK_ - 1);
                size_t idx;
                if (MODE == 0)
                    idx = (((size_t)b * H_ + h) * S_ + ss) * DK_ + d;
                else
                    idx = (((size_t)b * H_ + h) * DK_ + d) * S_ + ss;
                ((_Float16*)Y)[idx] = (_Float16)val;
            }
        }
    }
}

// ---------------------------------------------------------------------------
// Flash attention with TDM-staged, double-buffered K/V tiles in LDS.
//   Qh, Kh: [B][H][S][DK] f16;  Vt: [B][H][DK][S] f16 (transposed)
//   mask:   [B][S] int (masks QUERY rows per reference broadcast)
//   Ao:     [B][S][D] f16 concat output.
// Grid (S/64, H, B), block 128 = 4 waves; wave owns 16 query rows.
// ---------------------------------------------------------------------------
#define TW 72                                   // padded LDS row stride (halves)
#define KS_OFF(buf) ((buf) * 64 * TW)           // K tiles: 2 x 64 rows
#define VS_OFF(buf) ((2 + (buf)) * 64 * TW)     // V tiles: 2 x 64 rows
#define P_OFF(wv)   (4 * 64 * TW + (wv) * 16 * TW)
#define SMEM_HALVES (4 * 64 * TW + 4 * 16 * TW) // 46080 bytes

__global__ __launch_bounds__(128)
void mha_flash_attn_wmma_kernel(const _Float16* __restrict__ Qh,
                                const _Float16* __restrict__ Kh,
                                const _Float16* __restrict__ Vt,
                                const int* __restrict__ mask,
                                _Float16* __restrict__ Ao) {
    __shared__ _Float16 SMEM[SMEM_HALVES];

    const int tid  = threadIdx.x;
    const int lane = tid & 31;
    const int w    = tid >> 5;
    const int l15  = lane & 15;
    const int kc0  = (lane < 16) ? 0 : 8;
    const int hl   = (lane >= 16) ? 8 : 0;

    const int b  = blockIdx.z;
    const int h  = blockIdx.y;
    const int q0 = blockIdx.x * 64;
    const int qw = q0 + w * 16;

    const _Float16* Qb = Qh + ((size_t)b * H_ + h) * S_ * DK_;
    const _Float16* Kb = Kh + ((size_t)b * H_ + h) * S_ * DK_;
    const _Float16* Vb = Vt + ((size_t)b * H_ + h) * DK_ * S_;

    // ---- Q fragments resident in registers for the whole key loop ----
    v16h qf[2];
    #pragma unroll
    for (int ks = 0; ks < 2; ++ks) {
        const _Float16* p = Qb + (size_t)(qw + l15) * DK_ + ks * 32 + kc0;
        qf[ks] = cat8(*(const v8h*)p, *(const v8h*)(p + 16));
    }

    bool mok[8];
    float mrun[8], lrun[8];
    v8f o[4];
    #pragma unroll
    for (int i = 0; i < 8; ++i) {
        mok[i]  = mask[(size_t)b * S_ + qw + i + hl] != 0;
        mrun[i] = -3.0e38f;
        lrun[i] = 0.0f;
    }
    #pragma unroll
    for (int n = 0; n < 4; ++n) o[n] = (v8f){};

    // ---- TDM prologue: stage first K/V tiles into buffer 0 ----
    if (w == 0) {
        // K tile: rows = 64 keys, 64 contiguous dk halves each, row stride DK.
        tdm_load_tile_f16((unsigned)(KS_OFF(0) * 2), Kb, DK_, S_, 64, 64, DK_);
        // V tile: rows = 64 dk, 64 contiguous key halves each, row stride S.
        tdm_load_tile_f16((unsigned)(VS_OFF(0) * 2), Vb, S_, DK_, 64, 64, S_);
    }

    for (int kb = 0, it = 0; kb < S_; kb += 64, ++it) {
        const int cur = it & 1;
        if (w == 0) __builtin_amdgcn_s_wait_tensorcnt(0);
        __syncthreads();   // buffer `cur` visible to all waves

        // Prefetch next tiles into the other buffer while we compute.
        if (w == 0 && kb + 64 < S_) {
            tdm_load_tile_f16((unsigned)(KS_OFF(1 - cur) * 2),
                              Kb + (size_t)(kb + 64) * DK_, DK_, S_, 64, 64, DK_);
            tdm_load_tile_f16((unsigned)(VS_OFF(1 - cur) * 2),
                              Vb + (kb + 64), S_, DK_, 64, 64, S_);
        }

        const _Float16* Ks = SMEM + KS_OFF(cur);
        const _Float16* Vs = SMEM + VS_OFF(cur);
        _Float16* Pw = SMEM + P_OFF(w);

        // ---- scores: S = Q (16x64) @ K^T (64x64) -> 4 C-frags ----
        v8f sc[4];
        #pragma unroll
        for (int n = 0; n < 4; ++n) {
            v8f a = {};
            #pragma unroll
            for (int ks = 0; ks < 2; ++ks) {
                const _Float16* p = Ks + (n * 16 + l15) * TW + ks * 32 + kc0;
                v16h kf = cat8(*(const v8h*)p, *(const v8h*)(p + 16));
                a = wmma_f16(qf[ks], kf, a);
            }
            sc[n] = a;
        }

        // ---- online softmax (rows masked on QUERY dim per reference) ----
        #pragma unroll
        for (int i = 0; i < 8; ++i) {
            if (!mok[i]) {
                sc[0][i] = NEG_; sc[1][i] = NEG_;
                sc[2][i] = NEG_; sc[3][i] = NEG_;
            }
            float rm = fmaxf(fmaxf(sc[0][i], sc[1][i]), fmaxf(sc[2][i], sc[3][i]));
            rm = fmaxf(rm, __shfl_xor(rm, 1, 32));
            rm = fmaxf(rm, __shfl_xor(rm, 2, 32));
            rm = fmaxf(rm, __shfl_xor(rm, 4, 32));
            rm = fmaxf(rm, __shfl_xor(rm, 8, 32));
            const float nm    = fmaxf(mrun[i], rm);
            const float scale = __expf(mrun[i] - nm);
            mrun[i] = nm;

            float rsum = 0.0f;
            #pragma unroll
            for (int n = 0; n < 4; ++n) {
                const float pv = __expf(sc[n][i] - nm);
                sc[n][i] = pv;
                rsum += pv;
            }
            rsum += __shfl_xor(rsum, 1, 32);
            rsum += __shfl_xor(rsum, 2, 32);
            rsum += __shfl_xor(rsum, 4, 32);
            rsum += __shfl_xor(rsum, 8, 32);
            lrun[i] = lrun[i] * scale + rsum;

            #pragma unroll
            for (int n = 0; n < 4; ++n) o[n][i] *= scale;
        }

        // ---- P: C-layout -> A-layout via per-wave private LDS region ----
        #pragma unroll
        for (int n = 0; n < 4; ++n)
            #pragma unroll
            for (int i = 0; i < 8; ++i)
                Pw[(i + hl) * TW + n * 16 + l15] = (_Float16)sc[n][i];
        // wave-private RAW ordered by compiler-inserted s_wait_dscnt

        v16h pf[2];
        #pragma unroll
        for (int ks = 0; ks < 2; ++ks) {
            const _Float16* p = Pw + l15 * TW + ks * 32 + kc0;
            pf[ks] = cat8(*(const v8h*)p, *(const v8h*)(p + 16));
        }

        // ---- O += P (16x64) @ V (64x64), V^T tile rows = dk ----
        #pragma unroll
        for (int n = 0; n < 4; ++n) {
            #pragma unroll
            for (int ks = 0; ks < 2; ++ks) {
                const _Float16* p = Vs + (n * 16 + l15) * TW + ks * 32 + kc0;
                v16h vf = cat8(*(const v8h*)p, *(const v8h*)(p + 16));
                o[n] = wmma_f16(pf[ks], vf, o[n]);
            }
        }
        // Next iteration's top barrier orders the buffer swap: wave0 only
        // overwrites buffer `cur` after every wave has passed that barrier.
    }

    // ---- normalize + store concat output [B][S][D] ----
    #pragma unroll
    for (int i = 0; i < 8; ++i) {
        const float inv = 1.0f / lrun[i];
        const int q = qw + i + hl;
        #pragma unroll
        for (int n = 0; n < 4; ++n) {
            const int col = h * DK_ + n * 16 + l15;
            Ao[((size_t)b * S_ + q) * D_ + col] = (_Float16)(o[n][i] * inv);
        }
    }
}

// ---------------------------------------------------------------------------
extern "C" void kernel_launch(void* const* d_in, const int* in_sizes, int n_in,
                              void* d_out, int out_size, void* d_ws, size_t ws_size,
                              hipStream_t stream) {
    const float* q    = (const float*)d_in[0];
    const float* k    = (const float*)d_in[1];
    const float* v    = (const float*)d_in[2];
    const int*   mask = (const int*)d_in[3];
    const float* Wq   = (const float*)d_in[4];
    const float* bq   = (const float*)d_in[5];
    const float* Wk   = (const float*)d_in[6];
    const float* bk   = (const float*)d_in[7];
    const float* Wv   = (const float*)d_in[8];
    const float* bv   = (const float*)d_in[9];
    const float* Wo   = (const float*)d_in[10];
    const float* bo   = (const float*)d_in[11];
    float* out = (float*)d_out;

    char* ws = (char*)d_ws;
    const size_t headBytes = (size_t)B_ * H_ * S_ * DK_ * sizeof(_Float16); // 8 MB
    _Float16* Qh = (_Float16*)(ws);
    _Float16* Kh = (_Float16*)(ws + headBytes);
    _Float16* Vt = (_Float16*)(ws + 2 * headBytes);
    _Float16* Ao = (_Float16*)(ws + 3 * headBytes);

    dim3 pg((B_ * S_) / 64, D_ / 128);   // 64 x 8 blocks
    dim3 pb(256);

    mha_proj_wmma_kernel<0, 0><<<pg, pb, 0, stream>>>(q,  Wq, bq, (void*)Qh);
    mha_proj_wmma_kernel<0, 0><<<pg, pb, 0, stream>>>(k,  Wk, bk, (void*)Kh);
    mha_proj_wmma_kernel<0, 1><<<pg, pb, 0, stream>>>(v,  Wv, bv, (void*)Vt);

    mha_flash_attn_wmma_kernel<<<dim3(S_ / 64, H_, B_), 128, 0, stream>>>(
        Qh, Kh, Vt, mask, Ao);

    mha_proj_wmma_kernel<1, 2><<<pg, pb, 0, stream>>>(Ao, Wo, bo, (void*)out);
}
